// SimpleGraphEncoder_50508815401364
// MI455X (gfx1250) — compile-verified, matrix operand order checked by reading, not run on hardware
//
#include <hip/hip_runtime.h>
#include <hip/hip_bf16.h>

#define NNODE 512
#define NEDGE 130816   // 512*511/2
#define HID   64
#define BSZ   1024

typedef _Float16 v16h __attribute__((ext_vector_type(16)));
typedef _Float16 v8h  __attribute__((ext_vector_type(8)));
typedef float    v8f  __attribute__((ext_vector_type(8)));

// ---------------------------------------------------------------------------
// Kernel 1: degrees.  deg[b,n] = sum of adj[b,e] over edges incident to n.
// One workgroup (256 threads, 8 waves) per batch. Row-major upper-tri walk:
// row i is contiguous -> coalesced nontemporal loads; endpoint j handled by
// conflict-free per-wave LDS adds; endpoint i by a shfl-xor tree reduction.
// Per-wave private LDS slices keep float accumulation order deterministic.
// ---------------------------------------------------------------------------
__global__ __launch_bounds__(256) void deg_kernel(const float* __restrict__ adj,
                                                  float* __restrict__ deg) {
    const int b   = blockIdx.x;
    const int tid = threadIdx.x;
    const float* a = adj + (size_t)b * NEDGE;

    __shared__ float degl[8][NNODE];   // 16 KB: one slice per wave
    for (int i = tid; i < 8 * NNODE; i += 256) (&degl[0][0])[i] = 0.0f;
    __syncthreads();

    const int wave = tid >> 5;
    const int lane = tid & 31;

    for (int i = 0; i < NNODE - 1; ++i) {
        const int base = i * (NNODE - 1) - (i * (i - 1)) / 2;
        const int len  = NNODE - 1 - i;
        float acc = 0.0f;
        for (int k = tid; k < len; k += 256) {
            float v = __builtin_nontemporal_load(a + base + k);
            acc += v;
            // endpoint j = i+1+k : lane-distinct consecutive addresses,
            // single-writer slice -> deterministic, conflict-free
            atomicAdd(&degl[wave][i + 1 + k], v);
        }
        // endpoint i : reduce row-partial across the wave (fixed tree order)
        for (int off = 16; off > 0; off >>= 1) acc += __shfl_xor(acc, off, 32);
        if (lane == 0 && acc != 0.0f) atomicAdd(&degl[wave][i], acc);
    }
    __syncthreads();

    for (int n = tid; n < NNODE; n += 256) {
        float s = 0.0f;
        #pragma unroll
        for (int w = 0; w < 8; ++w) s += degl[w][n];   // fixed order
        deg[(size_t)b * NNODE + n] = s;
    }
}

// ---------------------------------------------------------------------------
// Kernel 2: g[b,:] = mean_n relu( relu(deg[b,n]*W1 + b1) @ W2 + b2 )
// One workgroup (8 waves) per batch; each wave owns 4 M-tiles of 16 nodes.
// A fragments (16x32 f16) are built in registers from deg + packed W1/b1;
// B fragments (32x16 f16 of W2^T) come from LDS (stride 72 halves: 16B
// aligned rows, bank-conflict-free for the 16-lane fragment pattern).
// D = v_wmma_f32_16x16x32_f16 chained over K; bias+ReLU on the f32 fragment,
// column sums accumulated for the node mean. Deterministic (per-wave slices).
// ---------------------------------------------------------------------------
__global__ __launch_bounds__(256) void mlp_wmma_kernel(
    const float* __restrict__ deg,
    const float* __restrict__ W1, const float* __restrict__ b1,
    const float* __restrict__ W2, const float* __restrict__ b2,
    float* __restrict__ g) {
    const int b    = blockIdx.x;
    const int tid  = threadIdx.x;
    const int lane = tid & 31;
    const int wave = tid >> 5;
    const int msub = lane & 15;
    const int hi   = lane >> 4;

    __shared__ __align__(16) _Float16 w2t[HID][72];  // W2 transposed, f16
    __shared__ float degl[NNODE];
    __shared__ float w1l[HID], b1l[HID], b2l[HID];
    __shared__ float gwl[8][HID];                    // per-wave partial sums

    for (int n = tid; n < NNODE; n += 256) degl[n] = deg[(size_t)b * NNODE + n];
    if (tid < HID) { w1l[tid] = W1[tid]; b1l[tid] = b1[tid]; b2l[tid] = b2[tid]; }
    for (int idx = tid; idx < HID * HID; idx += 256) {
        int n = idx & 63, k = idx >> 6;
        w2t[n][k] = (_Float16)W2[k * HID + n];       // w2t[n][k] = W2[k][n]
    }
    __syncthreads();

    union V16 { v16h v; v8h h[2]; };

    // Per-lane packed W1/b1 in fragment element order.
    // A-frag element e (kstep s): K = s*32 + hi*8 + (e<8 ? e : e+8)
    v16h w1f0, b1f0, w1f1, b1f1;
    #pragma unroll
    for (int e = 0; e < 16; ++e) {
        int k0 = hi * 8 + (e < 8 ? e : e + 8);
        w1f0[e] = (_Float16)w1l[k0];       b1f0[e] = (_Float16)b1l[k0];
        w1f1[e] = (_Float16)w1l[32 + k0];  b1f1[e] = (_Float16)b1l[32 + k0];
    }

    // Preload all 8 B fragments of W2 (4 N-tiles x 2 K-steps).
    v16h bf[4][2];
    #pragma unroll
    for (int nt = 0; nt < 4; ++nt)
        #pragma unroll
        for (int s = 0; s < 2; ++s) {
            int n  = nt * 16 + msub;
            int k0 = s * 32 + hi * 8;
            V16 u;
            u.h[0] = *(const v8h*)&w2t[n][k0];
            u.h[1] = *(const v8h*)&w2t[n][k0 + 16];
            bf[nt][s] = u.v;
        }

    v8f acc[4];
    #pragma unroll
    for (int nt = 0; nt < 4; ++nt)
        #pragma unroll
        for (int r = 0; r < 8; ++r) acc[nt][r] = 0.0f;

    const _Float16 h0 = (_Float16)0.0f;
    for (int q = 0; q < 4; ++q) {                 // 4 M-tiles per wave
        const int mt  = wave * 4 + q;
        const int row = mt * 16 + msub;
        const _Float16 dh = (_Float16)degl[row];
        v16h a0, a1;                              // h1 = relu(d*W1+b1), f16
        #pragma unroll
        for (int e = 0; e < 16; ++e) {
            _Float16 t0 = (_Float16)(dh * w1f0[e] + b1f0[e]);
            _Float16 t1 = (_Float16)(dh * w1f1[e] + b1f1[e]);
            a0[e] = t0 > h0 ? t0 : h0;
            a1[e] = t1 > h0 ? t1 : h0;
        }
        #pragma unroll
        for (int nt = 0; nt < 4; ++nt) {
            v8f c = {0.f, 0.f, 0.f, 0.f, 0.f, 0.f, 0.f, 0.f};
            c = __builtin_amdgcn_wmma_f32_16x16x32_f16(false, a0, false, bf[nt][0],
                                                       (short)0, c, false, false);
            c = __builtin_amdgcn_wmma_f32_16x16x32_f16(false, a1, false, bf[nt][1],
                                                       (short)0, c, false, false);
            const float bias = b2l[nt * 16 + msub];   // N = nt*16 + (lane&15)
            #pragma unroll
            for (int r = 0; r < 8; ++r) acc[nt][r] += fmaxf(c[r] + bias, 0.0f);
        }
    }

    // Column sums for the node mean: fold 8 VGPR rows, then the two M-halves
    // (lanes l and l^16 hold the same N). Lanes 0..15 write the wave slice.
    #pragma unroll
    for (int nt = 0; nt < 4; ++nt) {
        float s = 0.0f;
        #pragma unroll
        for (int r = 0; r < 8; ++r) s += acc[nt][r];
        s += __shfl_xor(s, 16, 32);
        if (lane < 16) gwl[wave][nt * 16 + lane] = s;
    }
    __syncthreads();

    if (tid < HID) {
        float s = 0.0f;
        #pragma unroll
        for (int w = 0; w < 8; ++w) s += gwl[w][tid];  // fixed order
        g[(size_t)b * HID + tid] = s * (1.0f / (float)NNODE);
    }
}

// ---------------------------------------------------------------------------
// Kernel 3: out[b,:] = relu(g[b,:] @ W3 + b3) @ W4 + b4   (tiny, exact f32)
// ---------------------------------------------------------------------------
__global__ __launch_bounds__(64) void head_kernel(
    const float* __restrict__ g,
    const float* __restrict__ W3, const float* __restrict__ b3,
    const float* __restrict__ W4, const float* __restrict__ b4,
    float* __restrict__ out) {
    const int b = blockIdx.x;
    const int o = threadIdx.x;
    __shared__ float gl[HID], t1[HID];
    gl[o] = g[(size_t)b * HID + o];
    __syncthreads();
    float a = b3[o];
    #pragma unroll 8
    for (int k = 0; k < HID; ++k) a = fmaf(gl[k], W3[k * HID + o], a);
    t1[o] = fmaxf(a, 0.0f);
    __syncthreads();
    float a2 = b4[o];
    #pragma unroll 8
    for (int k = 0; k < HID; ++k) a2 = fmaf(t1[k], W4[k * HID + o], a2);
    out[(size_t)b * HID + o] = a2;
}

// ---------------------------------------------------------------------------
extern "C" void kernel_launch(void* const* d_in, const int* in_sizes, int n_in,
                              void* d_out, int out_size, void* d_ws, size_t ws_size,
                              hipStream_t stream) {
    const float* adj = (const float*)d_in[0];
    const float* W1  = (const float*)d_in[1];
    const float* b1  = (const float*)d_in[2];
    const float* W2  = (const float*)d_in[3];
    const float* b2  = (const float*)d_in[4];
    const float* W3  = (const float*)d_in[5];
    const float* b3  = (const float*)d_in[6];
    const float* W4  = (const float*)d_in[7];
    const float* b4  = (const float*)d_in[8];
    float* out = (float*)d_out;

    float* deg = (float*)d_ws;                 // [BSZ, NNODE]  2 MB
    float* g   = deg + (size_t)BSZ * NNODE;    // [BSZ, HID]    256 KB

    deg_kernel<<<BSZ, 256, 0, stream>>>(adj, deg);
    mlp_wmma_kernel<<<BSZ, 256, 0, stream>>>(deg, W1, b1, W2, b2, g);
    head_kernel<<<BSZ, 64, 0, stream>>>(g, W3, b3, W4, b4, out);
}
